// DeepGMR_20143396618481
// MI455X (gfx1250) — matrix operand order repaired and checked by reading, not compiled, not run on hardware
//
#include <hip/hip_runtime.h>

#define NPTS   32768
#define BATCH  8
#define PTOT   (BATCH * NPTS)   // 262144 points per cloud
#define PB     128              // P tile per block
#define KS     32               // K step (bf16 WMMA K)

typedef __bf16 bf16_t;
typedef bf16_t v16bf __attribute__((ext_vector_type(16)));
typedef float  v8f   __attribute__((ext_vector_type(8)));
typedef unsigned int   u32;
typedef unsigned short u16;

// ---------------- bf16 helpers ----------------
__device__ __forceinline__ u16 f2bf(float f){
  union { float f; u32 u; } v; v.f = f;
  u32 r = v.u + 0x7FFFu + ((v.u >> 16) & 1u);   // round-to-nearest-even
  return (u16)(r >> 16);
}
__device__ __forceinline__ float bf2f(u16 h){
  union { u32 u; float f; } v; v.u = (u32)h << 16;
  return v.f;
}
// apply y = relu(x*sc+bi) to two packed bf16
__device__ __forceinline__ u32 affine2(u32 d, float sc, float bi){
  float lo = fmaxf(0.f, bf2f((u16)(d & 0xFFFFu)) * sc + bi);
  float hi = fmaxf(0.f, bf2f((u16)(d >> 16))     * sc + bi);
  return (u32)f2bf(lo) | ((u32)f2bf(hi) << 16);
}

union FragBF { v16bf v; u32 u[8]; uint4 q[2]; };

// ---------------- fused GEMM + BN-stats kernel ----------------
// Y[M,P] = W[M,K] @ act(X[K,P]); act = relu(x*inAff[k]+inAff[K+k]) if inAff.
// Adds per-batch bias (batchBias[b*M+m]) and per-channel bias (constBias[m]).
// Accumulates per-channel sum / sumsq into stats[2M] (pre-zeroed).
//
// LDS layout is FRAGMENT-MAJOR: each lane's 16-bf16 B fragment for N-tile t is
// 8 contiguous dwords at (t*256 + lane*8), so fragment reads are 2x ds_load_b128
// directly into the WMMA operand registers (no repack movs).
__global__ __launch_bounds__(256) void gemm_bn_kernel(
    const u16* __restrict__ W, const u16* __restrict__ X,
    u16* __restrict__ Y, float* __restrict__ Yf32,
    float* __restrict__ stats,
    const float* __restrict__ inAff,
    const float* __restrict__ batchBias,
    const float* __restrict__ constBias,
    int M, int K, int P)
{
  __shared__ u32 lds[2][(KS/2) * PB];   // 2 x 8KB
  const int tid  = threadIdx.x;
  const int lane = tid & 31;
  const int wave = tid >> 5;
  const int n0   = blockIdx.x * PB;
  const int mbase = blockIdx.y * 128 + wave * 16;
  const bool active = (mbase < M);
  const int half = lane >> 4;   // K-half selector
  const int col  = lane & 15;   // M row (A) / N col (B,D)

  v8f acc[8] = {};

  auto stage = [&](int ks, int buf){
    const int k0 = ks * KS;
    #pragma unroll
    for (int it = 0; it < 4; ++it){
      int tsk = tid + it * 256;           // 0..1023 tasks
      int kp  = tsk >> 6;                 // pair-row 0..15 (K pair {2kp, 2kp+1})
      int c2  = tsk & 63;                 // dword (2-col) index
      int k   = k0 + kp * 2;
      const u32* r0 = (const u32*)(X + (size_t)k * P + n0);
      const u32* r1 = (const u32*)(X + (size_t)(k + 1) * P + n0);
      u32 u0 = r0[c2], u1 = r1[c2];
      if (inAff){
        u0 = affine2(u0, inAff[k],     inAff[K + k]);
        u1 = affine2(u1, inAff[k + 1], inAff[K + k + 1]);
      }
      // transpose pairs: {k,k+1} packed per column
      u32 lo = (u0 & 0xFFFFu) | (u1 << 16);        // column 2*c2
      u32 hi = (u0 >> 16)     | (u1 & 0xFFFF0000u); // column 2*c2+1
      // fragment-major address: t*256 + (kp>>3)*128 + (n&15)*8 + (kp&7)
      int nA = 2 * c2, nB = 2 * c2 + 1;
      int hi7 = (kp >> 3) * 128 + (kp & 7);
      lds[buf][(nA >> 4) * 256 + (nA & 15) * 8 + hi7] = lo;
      lds[buf][(nB >> 4) * 256 + (nB & 15) * 8 + hi7] = hi;
    }
  };

  const int nk = K / KS;
  stage(0, 0);
  for (int ks = 0; ks < nk; ++ks){
    __syncthreads();
    if (ks + 1 < nk) stage(ks + 1, (ks + 1) & 1);
    if (active){
      // A fragment (16x32 bf16): two contiguous 16B chunks per lane
      FragBF a;
      const int m = mbase + col;
      const u32* wrow = (const u32*)(W + (size_t)m * K + ks * KS + 16 * half);
      a.q[0] = *(const uint4*)(wrow);       // VGPR 0-3: K pairs {0,2,4,6}+16h
      a.q[1] = *(const uint4*)(wrow + 4);   // VGPR 4-7: K pairs {8,10,12,14}+16h
      const u32* lb = lds[ks & 1];
      #pragma unroll
      for (int t = 0; t < 8; ++t){
        FragBF b;
        const u32* fb = lb + t * 256 + lane * 8;   // 32B contiguous, 32B aligned
        b.q[0] = *(const uint4*)(fb);
        b.q[1] = *(const uint4*)(fb + 4);
        acc[t] = __builtin_amdgcn_wmma_f32_16x16x32_bf16(
            false, a.v, false, b.v, (short)0, acc[t], false, false);
      }
    }
  }
  if (!active) return;

  const int bIdx = n0 / NPTS;   // PB divides NPTS -> single batch per block
  float rs[8] = {}, rq[8] = {};
  #pragma unroll
  for (int r = 0; r < 8; ++r){
    int m = mbase + r + 8 * half;          // D layout: VGPR r -> rows r / r+8
    float bias = 0.f;
    if (batchBias) bias += batchBias[bIdx * M + m];
    if (constBias) bias += constBias[m];
    #pragma unroll
    for (int t = 0; t < 8; ++t){
      float y = acc[t][r] + bias;
      rs[r] += y; rq[r] += y * y;
      size_t offo = (size_t)m * P + n0 + t * 16 + col;
      if (Y)    Y[offo]    = f2bf(y);
      if (Yf32) Yf32[offo] = y;
    }
  }
  #pragma unroll
  for (int r = 0; r < 8; ++r){
    float s = rs[r], q = rq[r];
    #pragma unroll
    for (int msk = 1; msk <= 8; msk <<= 1){
      s += __shfl_xor(s, msk, 32);
      q += __shfl_xor(q, msk, 32);
    }
    if (col == 0){
      int ch = mbase + r + 8 * half;
      atomicAdd(&stats[ch], s);
      atomicAdd(&stats[M + ch], q);
    }
  }
}

// ---------------- small kernels ----------------
__global__ void conv_w(const float* __restrict__ src, u16* __restrict__ dst,
                       int M, int Ksrc, int Koff, int Kcopy, int Kdst){
  int i = blockIdx.x * blockDim.x + threadIdx.x;
  if (i >= M * Kdst) return;
  int m = i / Kdst, k = i % Kdst;
  dst[i] = (k < Kcopy) ? f2bf(src[(size_t)m * Ksrc + Koff + k]) : (u16)0;
}

__global__ void zero_f32(float* p, int n){
  int i = blockIdx.x * blockDim.x + threadIdx.x;
  if (i < n) p[i] = 0.f;
}

// center coords per point, write padded [32,P] bf16
__global__ void prep_feats(const float* __restrict__ pts, u16* __restrict__ X0, int P){
  int p = blockIdx.x * blockDim.x + threadIdx.x;
  if (p >= P) return;
  float x = pts[(size_t)p * 3], y = pts[(size_t)p * 3 + 1], z = pts[(size_t)p * 3 + 2];
  float mn = (x + y + z) * (1.f / 3.f);
  X0[p]               = f2bf(x - mn);
  X0[(size_t)P + p]   = f2bf(y - mn);
  X0[2*(size_t)P + p] = f2bf(z - mn);
  for (int k = 3; k < 32; ++k) X0[(size_t)k * P + p] = 0;
}

__global__ void bn_finalize(const float* __restrict__ stats, const float* __restrict__ g,
                            const float* __restrict__ b, float* __restrict__ aff,
                            int M, float invP){
  int m = blockIdx.x * blockDim.x + threadIdx.x;
  if (m >= M) return;
  float mean = stats[m] * invP;
  float var  = stats[M + m] * invP - mean * mean;
  float sc   = g[m] * rsqrtf(var + 1e-5f);
  aff[m]     = sc;
  aff[M + m] = b[m] - mean * sc;
}

// per (batch, channel) max over N of activated f
__global__ void maxpool_k(const u16* __restrict__ F, const float* __restrict__ aff,
                          float* __restrict__ out, int C, int P){
  int c = blockIdx.x, b = blockIdx.y, tid = threadIdx.x;
  float sc = aff[c], bi = aff[C + c];
  const u16* row = F + (size_t)c * P + (size_t)b * NPTS;
  float mx = -1e30f;
  for (int n = tid; n < NPTS; n += blockDim.x)
    mx = fmaxf(mx, fmaxf(0.f, bf2f(row[n]) * sc + bi));
  __shared__ float sh[256];
  sh[tid] = mx; __syncthreads();
  for (int s = 128; s > 0; s >>= 1){
    if (tid < s) sh[tid] = fmaxf(sh[tid], sh[tid + s]);
    __syncthreads();
  }
  if (tid == 0) out[b * C + c] = sh[0];
}

// dec1 global-feature bias: bb[b,o] = sum_c Wd[o,1024][512+c] * fg[b,c]
__global__ void dec1_bias(const float* __restrict__ Wd, const float* __restrict__ fg,
                          float* __restrict__ bb){
  int i = blockIdx.x * blockDim.x + threadIdx.x;
  if (i >= BATCH * 512) return;
  int b = i >> 9, o = i & 511;
  const float* w = Wd + (size_t)o * 1024 + 512;
  const float* f = fg + b * 512;
  float s = 0.f;
  for (int c = 0; c < 512; ++c) s += w[c] * f[c];
  bb[i] = s;
}

__global__ void softmax_k(const float* __restrict__ logits, float* __restrict__ gamma, int P){
  int p = blockIdx.x * blockDim.x + threadIdx.x;
  if (p >= P) return;
  float v[16], mx = -1e30f;
  #pragma unroll
  for (int j = 0; j < 16; ++j){ v[j] = logits[(size_t)j * P + p]; mx = fmaxf(mx, v[j]); }
  float s = 0.f;
  #pragma unroll
  for (int j = 0; j < 16; ++j){ v[j] = __expf(v[j] - mx); s += v[j]; }
  float inv = 1.f / s;
  #pragma unroll
  for (int j = 0; j < 16; ++j) gamma[(size_t)p * 16 + j] = v[j] * inv;
}

// per-batch moments: S0[16], S1[16*3], S2[16]  (80 floats / batch)
__global__ void gmm_accum(const float* __restrict__ pts, const float* __restrict__ gamma,
                          float* __restrict__ S, int npts){
  __shared__ float sh[80];
  int tid = threadIdx.x;
  for (int i = tid; i < 80; i += blockDim.x) sh[i] = 0.f;
  __syncthreads();
  int p = blockIdx.x * blockDim.x + tid;
  int b = p / npts;
  float x = pts[(size_t)p*3], y = pts[(size_t)p*3+1], z = pts[(size_t)p*3+2];
  float r2 = x*x + y*y + z*z;
  const float* g = gamma + (size_t)p * 16;
  #pragma unroll
  for (int j = 0; j < 16; ++j){
    float gj = g[j];
    atomicAdd(&sh[j], gj);
    atomicAdd(&sh[16 + j*3 + 0], gj * x);
    atomicAdd(&sh[16 + j*3 + 1], gj * y);
    atomicAdd(&sh[16 + j*3 + 2], gj * z);
    atomicAdd(&sh[64 + j], gj * r2);
  }
  __syncthreads();
  float* Sb = S + (size_t)b * 80;
  for (int i = tid; i < 80; i += blockDim.x) atomicAdd(&Sb[i], sh[i]);
}

// prm[b][j] = {pi, mu0, mu1, mu2, sigma_iso}
__global__ void gmm_fin(const float* __restrict__ S, float* __restrict__ prm, int npts){
  int t = threadIdx.x;
  if (t >= BATCH * 16) return;
  int b = t >> 4, j = t & 15;
  const float* s = S + b * 80;
  float S0 = s[j], inv = 1.f / S0;
  float m0 = s[16+j*3]*inv, m1 = s[16+j*3+1]*inv, m2 = s[16+j*3+2]*inv;
  float var = s[64+j]*inv - (m0*m0 + m1*m1 + m2*m2);   // S2/S0 - |mu|^2
  float* o = prm + t * 5;
  o[0] = S0 / (float)npts; o[1] = m0; o[2] = m1; o[3] = m2; o[4] = var;
}

// ---------------- registration (3x3 SVD via Jacobi) ----------------
__device__ __forceinline__ float det3(const float* A){
  return A[0]*(A[4]*A[8]-A[5]*A[7]) - A[1]*(A[3]*A[8]-A[5]*A[6]) + A[2]*(A[3]*A[7]-A[4]*A[6]);
}

__device__ void svd_rot(const float Ms[9], float R[9]){
  float A[9], V[9] = {1,0,0, 0,1,0, 0,0,1};
  for (int a = 0; a < 3; ++a)
    for (int c = 0; c < 3; ++c){
      float s = 0.f;
      for (int k = 0; k < 3; ++k) s += Ms[k*3+a] * Ms[k*3+c];
      A[a*3+c] = s;                       // Ms^T Ms
    }
  const int PQ[3][2] = {{0,1},{0,2},{1,2}};
  for (int sweep = 0; sweep < 12; ++sweep)
    for (int pi = 0; pi < 3; ++pi){
      int p = PQ[pi][0], q = PQ[pi][1];
      float apq = A[p*3+q];
      if (fabsf(apq) < 1e-20f) continue;
      float tau = (A[q*3+q] - A[p*3+p]) / (2.f * apq);
      float tt  = (tau >= 0.f ? 1.f : -1.f) / (fabsf(tau) + sqrtf(1.f + tau*tau));
      float c   = rsqrtf(1.f + tt*tt), s = tt * c;
      for (int k = 0; k < 3; ++k){
        float akp = A[k*3+p], akq = A[k*3+q];
        A[k*3+p] = c*akp - s*akq; A[k*3+q] = s*akp + c*akq;
      }
      for (int k = 0; k < 3; ++k){
        float apk = A[p*3+k], aqk = A[q*3+k];
        A[p*3+k] = c*apk - s*aqk; A[q*3+k] = s*apk + c*aqk;
      }
      for (int k = 0; k < 3; ++k){
        float vkp = V[k*3+p], vkq = V[k*3+q];
        V[k*3+p] = c*vkp - s*vkq; V[k*3+q] = s*vkp + c*vkq;
      }
    }
  float w[3] = {A[0], A[4], A[8]};
  for (int i = 0; i < 2; ++i)
    for (int k = i+1; k < 3; ++k)
      if (w[k] > w[i]){
        float tw = w[i]; w[i] = w[k]; w[k] = tw;
        for (int r = 0; r < 3; ++r){ float tv = V[r*3+i]; V[r*3+i] = V[r*3+k]; V[r*3+k] = tv; }
      }
  float U[9];
  for (int i = 0; i < 3; ++i){
    float sg = sqrtf(fmaxf(w[i], 0.f));
    float inv = sg > 1e-12f ? 1.f / sg : 0.f;
    for (int r = 0; r < 3; ++r)
      U[r*3+i] = (Ms[r*3]*V[i] + Ms[r*3+1]*V[3+i] + Ms[r*3+2]*V[6+i]) * inv;
  }
  float VUt[9];
  for (int a = 0; a < 3; ++a)
    for (int c = 0; c < 3; ++c)
      VUt[a*3+c] = V[a*3]*U[c*3] + V[a*3+1]*U[c*3+1] + V[a*3+2]*U[c*3+2];
  float d = det3(VUt);
  for (int a = 0; a < 3; ++a)
    for (int c = 0; c < 3; ++c)
      R[a*3+c] = V[a*3]*U[c*3] + V[a*3+1]*U[c*3+1] + d * V[a*3+2]*U[c*3+2];
}

__global__ void reg_kernel(const float* __restrict__ prm1, const float* __restrict__ prm2,
                           float* __restrict__ T12, float* __restrict__ T21){
  int t = threadIdx.x;
  if (t >= 16) return;
  int b = t & 7;
  const float* Ps = (t < 8) ? prm1 + b*80 : prm2 + b*80;
  const float* Pt = (t < 8) ? prm2 + b*80 : prm1 + b*80;
  float* T = ((t < 8) ? T12 : T21) + b * 16;
  float cs[3] = {0,0,0}, ct[3] = {0,0,0};
  for (int j = 0; j < 16; ++j){
    float pi = Ps[j*5];
    for (int d = 0; d < 3; ++d){ cs[d] += pi * Ps[j*5+1+d]; ct[d] += pi * Pt[j*5+1+d]; }
  }
  float Ms[9] = {};
  for (int j = 0; j < 16; ++j){
    float w = Ps[j*5] / Pt[j*5+4];    // pi / sigma_t (isotropic inverse)
    float ds[3], dt[3];
    for (int d = 0; d < 3; ++d){ ds[d] = Ps[j*5+1+d] - cs[d]; dt[d] = Pt[j*5+1+d] - ct[d]; }
    for (int a = 0; a < 3; ++a)
      for (int c = 0; c < 3; ++c) Ms[a*3+c] += w * ds[a] * dt[c];
  }
  float R[9];
  svd_rot(Ms, R);
  for (int a = 0; a < 3; ++a){
    T[a*4+0] = R[a*3+0]; T[a*4+1] = R[a*3+1]; T[a*4+2] = R[a*3+2];
    T[a*4+3] = ct[a] - (R[a*3]*cs[0] + R[a*3+1]*cs[1] + R[a*3+2]*cs[2]);
  }
  T[12] = 0.f; T[13] = 0.f; T[14] = 0.f; T[15] = 1.f;
}

// ---------------- loss / error metrics ----------------
__global__ void loss_kernel(const float* __restrict__ T12, const float* __restrict__ T21,
                            const float* __restrict__ Tgt, const float* __restrict__ pts1,
                            float* __restrict__ out){
  __shared__ float sh[16];
  int t = threadIdx.x;
  if (t < 8){
    int b = t;
    const float* A  = T12 + b*16;
    const float* Bm = T21 + b*16;
    const float* G  = Tgt + b*16;
    float Gi[16];
    for (int r = 0; r < 3; ++r)
      for (int c = 0; c < 3; ++c) Gi[r*4+c] = G[c*4+r];
    for (int r = 0; r < 3; ++r)
      Gi[r*4+3] = -(G[r]*G[3] + G[4+r]*G[7] + G[8+r]*G[11]);
    Gi[12] = 0.f; Gi[13] = 0.f; Gi[14] = 0.f; Gi[15] = 1.f;
    float e1 = 0.f, e2 = 0.f;
    for (int r = 0; r < 4; ++r)
      for (int c = 0; c < 4; ++c){
        float m1 = 0.f, m2 = 0.f;
        for (int k = 0; k < 4; ++k){ m1 += A[r*4+k]*Gi[k*4+c]; m2 += Bm[r*4+k]*G[k*4+c]; }
        m1 -= (r == c) ? 1.f : 0.f; m2 -= (r == c) ? 1.f : 0.f;
        e1 += m1*m1; e2 += m2*m2;
      }
    sh[b] = e1; sh[8+b] = e2;
    float acc = 0.f;
    for (int i = 0; i < 100; ++i){
      const float* p = pts1 + ((size_t)b * NPTS + i) * 3;
      float d2 = 0.f;
      for (int r = 0; r < 3; ++r){
        float pp = A[r*4]*p[0] + A[r*4+1]*p[1] + A[r*4+2]*p[2] + A[r*4+3];
        float pg = G[r*4]*p[0] + G[r*4+1]*p[1] + G[r*4+2]*p[2] + G[r*4+3];
        float d = pp - pg; d2 += d*d;
      }
      acc += sqrtf(d2);
    }
    out[7 + b] = acc * 0.01f;
    if (b < 3){
      float dot = 0.f;
      for (int r = 0; r < 3; ++r)
        for (int c = 0; c < 4; ++c) dot += A[r*4+c] * G[r*4+c];
      float cs = fminf(1.f, fmaxf(-1.f, (dot - 1.f) * 0.5f));
      out[1 + b] = acosf(cs) * 57.29577951308232f;
      float te = 0.f;
      for (int c = 0; c < 4; ++c){ float d = A[12+c] - G[12+c]; te += d*d; }
      out[4 + b] = sqrtf(te);
    }
  }
  __syncthreads();
  if (t == 0){
    float s = 0.f;
    for (int i = 0; i < 16; ++i) s += sh[i];
    out[0] = s / 128.f;   // (sum e1 + sum e2) / (B*16)
  }
}

// ---------------- orchestration ----------------
extern "C" void kernel_launch(void* const* d_in, const int* in_sizes, int n_in,
                              void* d_out, int out_size, void* d_ws, size_t ws_size,
                              hipStream_t stream){
  (void)in_sizes; (void)n_in; (void)out_size; (void)ws_size;
  const int P = PTOT;
  char* base = (char*)d_ws;
  size_t off = 0;
  auto alloc = [&](size_t bytes)->void*{
    off = (off + 255) & ~(size_t)255;
    void* p = base + off; off += bytes; return p;
  };

  u16* buf0 = (u16*)alloc((size_t)512 * P * 2);
  u16* buf1 = (u16*)alloc((size_t)512 * P * 2);
  float* logits = (float*)alloc((size_t)16 * P * 4);
  float* gamma  = (float*)alloc((size_t)P * 16 * 4);
  u16* wE1 = (u16*)alloc((size_t)64  * 32  * 2);
  u16* wE2 = (u16*)alloc((size_t)128 * 64  * 2);
  u16* wE3 = (u16*)alloc((size_t)256 * 128 * 2);
  u16* wE4 = (u16*)alloc((size_t)512 * 256 * 2);
  u16* wD1 = (u16*)alloc((size_t)512 * 512 * 2);
  u16* wD2 = (u16*)alloc((size_t)256 * 512 * 2);
  u16* wD3 = (u16*)alloc((size_t)128 * 256 * 2);
  u16* wL  = (u16*)alloc((size_t)16  * 128 * 2);
  float* stats = (float*)alloc(1024 * 4);
  float* affA  = (float*)alloc(1024 * 4);
  float* affB  = (float*)alloc(1024 * 4);
  float* fglob = (float*)alloc(BATCH * 512 * 4);
  float* bbias = (float*)alloc(BATCH * 512 * 4);
  float* gmmS  = (float*)alloc(BATCH * 80 * 4);
  float* prm0  = (float*)alloc(BATCH * 80 * 4);
  float* prm1p = (float*)alloc(BATCH * 80 * 4);
  float* T12b  = (float*)alloc(BATCH * 16 * 4);
  float* T21b  = (float*)alloc(BATCH * 16 * 4);

  auto cw = [&](int inIdx, u16* dst, int M, int Ksrc, int Koff, int Kcopy, int Kdst){
    int n = M * Kdst;
    conv_w<<<(n + 255) / 256, 256, 0, stream>>>((const float*)d_in[inIdx], dst, M, Ksrc, Koff, Kcopy, Kdst);
  };
  cw(3,  wE1, 64,  3,    0, 3,   32);
  cw(4,  wE2, 128, 64,   0, 64,  64);
  cw(5,  wE3, 256, 128,  0, 128, 128);
  cw(6,  wE4, 512, 256,  0, 256, 256);
  cw(15, wD1, 512, 1024, 0, 512, 512);
  cw(16, wD2, 256, 512,  0, 512, 512);
  cw(17, wD3, 128, 256,  0, 256, 256);
  cw(24, wL,  16,  128,  0, 128, 128);

  const float invP = 1.f / (float)P;
  float* prms[2] = { prm0, prm1p };

  auto rungemm = [&](const u16* W, const u16* X, u16* Y, float* Yf,
                     const float* inA, const float* bb2, const float* cb, int M, int K){
    zero_f32<<<(2 * M + 255) / 256, 256, 0, stream>>>(stats, 2 * M);
    dim3 g(P / PB, (M + 127) / 128);
    gemm_bn_kernel<<<g, 256, 0, stream>>>(W, X, Y, Yf, stats, inA, bb2, cb, M, K, P);
  };

  for (int c = 0; c < 2; ++c){
    const float* pts = (const float*)d_in[c];
    prep_feats<<<P / 256, 256, 0, stream>>>(pts, buf0, P);
    // encoder
    rungemm(wE1, buf0, buf1, nullptr, nullptr, nullptr, nullptr, 64, 32);
    bn_finalize<<<1, 64, 0, stream>>>(stats, (const float*)d_in[7], (const float*)d_in[11], affA, 64, invP);
    rungemm(wE2, buf1, buf0, nullptr, affA, nullptr, nullptr, 128, 64);
    bn_finalize<<<1, 128, 0, stream>>>(stats, (const float*)d_in[8], (const float*)d_in[12], affB, 128, invP);
    rungemm(wE3, buf0, buf1, nullptr, affB, nullptr, nullptr, 256, 128);
    bn_finalize<<<1, 256, 0, stream>>>(stats, (const float*)d_in[9], (const float*)d_in[13], affA, 256, invP);
    rungemm(wE4, buf1, buf0, nullptr, affA, nullptr, nullptr, 512, 256);
    bn_finalize<<<2, 256, 0, stream>>>(stats, (const float*)d_in[10], (const float*)d_in[14], affB, 512, invP);
    // global max + dec1 per-batch bias
    maxpool_k<<<dim3(512, BATCH), 256, 0, stream>>>(buf0, affB, fglob, 512, P);
    dec1_bias<<<(BATCH * 512) / 256, 256, 0, stream>>>((const float*)d_in[15], fglob, bbias);
    // decoder
    rungemm(wD1, buf0, buf1, nullptr, affB, bbias, nullptr, 512, 512);
    bn_finalize<<<2, 256, 0, stream>>>(stats, (const float*)d_in[18], (const float*)d_in[21], affA, 512, invP);
    rungemm(wD2, buf1, buf0, nullptr, affA, nullptr, nullptr, 256, 512);
    bn_finalize<<<1, 256, 0, stream>>>(stats, (const float*)d_in[19], (const float*)d_in[22], affB, 256, invP);
    rungemm(wD3, buf0, buf1, nullptr, affB, nullptr, nullptr, 128, 256);
    bn_finalize<<<1, 128, 0, stream>>>(stats, (const float*)d_in[20], (const float*)d_in[23], affA, 128, invP);
    rungemm(wL, buf1, nullptr, logits, affA, nullptr, (const float*)d_in[25], 16, 128);
    // softmax + GMM moments
    softmax_k<<<P / 256, 256, 0, stream>>>(logits, gamma, P);
    zero_f32<<<3, 256, 0, stream>>>(gmmS, BATCH * 80);
    gmm_accum<<<P / 256, 256, 0, stream>>>(pts, gamma, gmmS, NPTS);
    gmm_fin<<<1, 128, 0, stream>>>(gmmS, prms[c], NPTS);
  }

  reg_kernel<<<1, 32, 0, stream>>>(prm0, prm1p, T12b, T21b);
  loss_kernel<<<1, 64, 0, stream>>>(T12b, T21b, (const float*)d_in[2], (const float*)d_in[0], (float*)d_out);
}
